// HALTModel_51745765982328
// MI455X (gfx1250) — matrix operand order repaired and checked by reading, not compiled, not run on hardware
//
#include <hip/hip_runtime.h>
#include <math.h>

// ---------------- problem constants ----------------
#define BB   64
#define TT   1024
#define KK   20
#define FEAT 25
#define KPAD 32
#define PROJ 512
#define HH   256
#define GG   768          // 3*H
#define MROWS (BB*TT)     // 65536
#define EPSC 1e-9f

typedef _Float16 v16h __attribute__((ext_vector_type(16)));
typedef _Float16 v8h  __attribute__((ext_vector_type(8)));
typedef float    v8f  __attribute__((ext_vector_type(8)));

__device__ __forceinline__ float sigm(float x){ return 1.0f/(1.0f+expf(-x)); }
__device__ __forceinline__ float gelu_exact(float x){ return 0.5f*x*(1.0f+erff(x*0.70710678118654752f)); }

// h_dec from sel/best_alt
__device__ __forceinline__ float hdec_from(float sel, float ba){
    float pc = sigm(sel - ba); // softmax([sel,ba])[0]
    return -(pc*logf(pc+EPSC) + (1.0f-pc)*logf(1.0f-pc+EPSC));
}
__device__ __forceinline__ float hdec_of(const float* lp){
    float sel = lp[0];
    float ba = lp[1];
    #pragma unroll
    for (int k=2;k<KK;k++) ba = fmaxf(ba, lp[k]);
    return hdec_from(sel, ba);
}

// ---------------- kernel 1: features + layernorm -> f16 (M x 32, zero padded) --------
__global__ void features_ln(const float* __restrict__ raw, const float* __restrict__ ln_g,
                            const float* __restrict__ ln_b, _Float16* __restrict__ X){
    int idx = blockIdx.x*blockDim.x + threadIdx.x;   // 0..65535
    if (idx >= MROWS) return;
    int t = idx & (TT-1);
    const float* lp = raw + (size_t)idx*KK;

    float f[FEAT];
    float m = lp[0];
    #pragma unroll
    for (int k=1;k<KK;k++) m = fmaxf(m, lp[k]);
    float Z = 0.0f, avg = 0.0f;
    #pragma unroll
    for (int k=0;k<KK;k++){ Z += expf(lp[k]-m); avg += lp[k]; }
    avg *= (1.0f/KK);
    float hov = 0.0f;
    #pragma unroll
    for (int k=0;k<KK;k++){ float p = expf(lp[k]-m)/Z; hov -= p*logf(p+EPSC); }

    float sel = lp[0];
    float ma = lp[1];
    int rank = 1;
    #pragma unroll
    for (int k=1;k<KK;k++){ ma = fmaxf(ma, lp[k]); if (lp[k] > sel) rank++; }
    float Za = 0.0f;
    #pragma unroll
    for (int k=1;k<KK;k++) Za += expf(lp[k]-ma);
    float ha = 0.0f;
    #pragma unroll
    for (int k=1;k<KK;k++){ float p = expf(lp[k]-ma)/Za; ha -= p*logf(p+EPSC); }

    float hd  = hdec_from(sel, ma);
    float hdp = (t==0) ? hd : hdec_of(lp - KK);
    float dh  = hd - hdp;

    #pragma unroll
    for (int k=0;k<KK;k++) f[k] = lp[k];
    f[20]=avg; f[21]=(float)rank; f[22]=hov; f[23]=ha; f[24]=dh;

    float mu=0.0f;
    #pragma unroll
    for (int c=0;c<FEAT;c++) mu += f[c];
    mu *= (1.0f/FEAT);
    float var=0.0f;
    #pragma unroll
    for (int c=0;c<FEAT;c++){ float d=f[c]-mu; var += d*d; }
    var *= (1.0f/FEAT);
    float inv = rsqrtf(var + 1e-5f);

    _Float16* xr = X + (size_t)idx*KPAD;
    #pragma unroll
    for (int c=0;c<FEAT;c++) xr[c] = (_Float16)((f[c]-mu)*inv*ln_g[c] + ln_b[c]);
    #pragma unroll
    for (int c=FEAT;c<KPAD;c++) xr[c] = (_Float16)0.0f;
}

// ---------------- kernel 2: f32 -> f16 weight convert with K-padding -----------------
__global__ void convert_pad(const float* __restrict__ src, _Float16* __restrict__ dst,
                            int rows, int scols, int dcols){
    int i = blockIdx.x*blockDim.x + threadIdx.x;
    int total = rows*dcols;
    if (i >= total) return;
    int r = i / dcols, c = i - r*dcols;
    dst[i] = (c < scols) ? (_Float16)src[(size_t)r*scols + c] : (_Float16)0.0f;
}

// ---------------- kernel 3: WMMA GEMM  C[m,n] = act(sum_k A[m,k]*Bw[n,k] + bias[n]) ---
// A: MxK f16 row-major, Bw: NxK f16 row-major (weights, i.e. B = Bw^T).
// One wave per 16x16 output tile; 4 waves / block cover 64 output columns.
// All operands are L2-resident (192MB) -> direct global fragment loads.
__global__ void gemm_wmma(const _Float16* __restrict__ A, const _Float16* __restrict__ Bw,
                          const float* __restrict__ bias, float* __restrict__ outF,
                          _Float16* __restrict__ outH, int M, int N, int K,
                          int act, const int* __restrict__ lengths, int T){
    int wid  = threadIdx.x >> 5;
    int lane = threadIdx.x & 31;
    int n0 = (blockIdx.x*4 + wid) * 16;
    int m0 = blockIdx.y * 16;
    int rowA = m0 + (lane & 15);
    int colB = n0 + (lane & 15);
    int kA = (lane & 16) ? 16 : 0;   // A: lanes16-31 hold K=16..31
    int kB = (lane & 16) ? 16 : 0;   // B: lanes16-31 hold K=16..31
    const _Float16* ap = A  + (size_t)rowA*K + kA;
    const _Float16* bp = Bw + (size_t)colB*K + kB;

    v8f c = {};
    for (int k0 = 0; k0 < K; k0 += 32){
        if (k0 + 32 < K) __builtin_prefetch(bp + k0 + 32, 0, 1);
        v8h lo = *(const v8h*)(ap + k0);
        v8h hi = *(const v8h*)(ap + k0 + 8);
        v16h a = __builtin_shufflevector(lo, hi, 0,1,2,3,4,5,6,7,8,9,10,11,12,13,14,15);
        v16h b = *(const v16h*)(bp + k0);
        c = __builtin_amdgcn_wmma_f32_16x16x32_f16(false, a, false, b, (short)0, c, false, false);
    }

    float bv = bias[colB];
    int rbase = m0 + ((lane & 16) ? 8 : 0);
    #pragma unroll
    for (int i=0;i<8;i++){
        int r = rbase + i;
        float v = c[i] + bv;
        if (act) v = gelu_exact(v);
        if (lengths){
            int bb = r / T; int tt = r - bb*T;
            if (tt >= lengths[bb]) v = 0.0f;
        }
        size_t oi = (size_t)r*N + colB;
        if (outF) outF[oi] = v;
        if (outH) outH[oi] = (_Float16)v;
    }
}

// ---------------- kernel 4: GRU recurrent scan (one block per direction) -------------
// LDS: ph (64x768 f32, 192KB) + h32 (64x256 f32, 64KB) + h16 (64x256 f16, 32KB) = 288KB
#define SMEM_PH   (BB*GG*4)
#define SMEM_H32  (BB*HH*4)
#define SMEM_H16  (BB*HH*2)
#define SMEM_GRU  (SMEM_PH + SMEM_H32 + SMEM_H16)

__global__ void gru_scan(const float* __restrict__ preF, const float* __restrict__ preB,
                         const _Float16* __restrict__ WhhF, const _Float16* __restrict__ WhhB,
                         const float* __restrict__ bihF, const float* __restrict__ bihB,
                         const float* __restrict__ bhhF, const float* __restrict__ bhhB,
                         float* __restrict__ ysF, float* __restrict__ ysB,
                         const int* __restrict__ lengths){
    extern __shared__ char smem[];
    float*    ph  = (float*)smem;
    float*    h32 = (float*)(smem + SMEM_PH);
    _Float16* h16 = (_Float16*)(smem + SMEM_PH + SMEM_H32);

    const int dir = blockIdx.x;
    const float*    pre  = dir ? preB : preF;
    const _Float16* Whh  = dir ? WhhB : WhhF;
    const float*    bih  = dir ? bihB : bihF;
    const float*    bhh  = dir ? bhhB : bhhF;
    float*          ys   = dir ? ysB  : ysF;

    const int tid  = threadIdx.x;
    const int wid  = tid >> 5;
    const int lane = tid & 31;

    for (int e = tid; e < BB*HH; e += 256){ h32[e] = 0.0f; h16[e] = (_Float16)0.0f; }
    __syncthreads();

    // per-wave tile geometry (uniform across waves -> EXEC all ones at WMMA)
    const int kA = (lane & 16) ? 16 : 0;
    const int kB = (lane & 16) ? 16 : 0;
    const int l15 = lane & 15;
    const int rsel = (lane & 16) ? 8 : 0;

    for (int t = 0; t < TT; ++t){
        // ---- phase 1: ph = h @ Whh^T  (64x768, K=256) via 192 WMMA tiles ----
        for (int tile = wid; tile < (BB/16)*(GG/16); tile += 8){
            int mt = tile & 3;          // 4 M-tiles (batch)
            int nt = tile >> 2;         // 48 N-tiles (gates)
            int m0 = mt*16, n0 = nt*16;
            int row = m0 + l15;
            int col = n0 + l15;
            const _Float16* hp = h16 + (size_t)row*HH + kA;
            const _Float16* wp = Whh + (size_t)col*HH + kB;
            v8f c = {};
            #pragma unroll
            for (int k0 = 0; k0 < HH; k0 += 32){
                v8h lo = *(const v8h*)(hp + k0);
                v8h hi = *(const v8h*)(hp + k0 + 8);
                v16h a = __builtin_shufflevector(lo, hi, 0,1,2,3,4,5,6,7,8,9,10,11,12,13,14,15);
                v16h b = *(const v16h*)(wp + k0);
                c = __builtin_amdgcn_wmma_f32_16x16x32_f16(false, a, false, b, (short)0, c, false, false);
            }
            int rbase = m0 + rsel;
            #pragma unroll
            for (int i=0;i<8;i++) ph[(size_t)(rbase+i)*GG + col] = c[i];
        }
        __syncthreads();

        // ---- phase 2: gates, state update, output scatter ----
        for (int e = tid; e < BB*HH; e += 256){
            int b  = e >> 8;
            int hh = e & (HH-1);
            int len = lengths[b];
            float irv, izv, innv;
            if (dir == 0){
                const float* pr = pre + ((size_t)(b*TT + t))*GG;
                irv = pr[hh]; izv = pr[HH+hh]; innv = pr[2*HH+hh];
            } else {
                int tm = len - 1 - t;
                if (tm >= 0){
                    const float* pr = pre + ((size_t)(b*TT + tm))*GG;
                    irv = pr[hh]; izv = pr[HH+hh]; innv = pr[2*HH+hh];
                } else {      // reversed-padding region: x == 0 -> pre == bih
                    irv = bih[hh]; izv = bih[HH+hh]; innv = bih[2*HH+hh];
                }
            }
            float hr = ph[(size_t)b*GG + hh]        + bhh[hh];
            float hz = ph[(size_t)b*GG + HH + hh]   + bhh[HH+hh];
            float hn = ph[(size_t)b*GG + 2*HH + hh] + bhh[2*HH+hh];
            float r = sigm(irv + hr);
            float z = sigm(izv + hz);
            float n = tanhf(innv + r*hn);
            float hold = h32[e];
            float hnew = (1.0f - z)*n + z*hold;
            h32[e] = hnew;
            h16[e] = (_Float16)hnew;
            if (dir == 0){
                ys[((size_t)(b*TT + t))*HH + hh] = hnew;
            } else if (t < len){   // double-reversal: scan step t lands at position len-1-t
                ys[((size_t)(b*TT + (len-1-t)))*HH + hh] = hnew;
            }
        }
        __syncthreads();
    }
}

// ---------------- kernel 5: concat fwd/bwd + mask -> f16 (M x 512) -------------------
__global__ void concat_mask(const float* __restrict__ ysF, const float* __restrict__ ysB,
                            const int* __restrict__ lengths, _Float16* __restrict__ out){
    size_t i = (size_t)blockIdx.x*blockDim.x + threadIdx.x;
    if (i >= (size_t)MROWS*(2*HH)) return;
    int r = (int)(i >> 9);
    int c = (int)(i & (2*HH - 1));
    int b = r >> 10;
    int t = r & (TT-1);
    float v;
    if (t < lengths[b])
        v = (c < HH) ? ysF[(size_t)r*HH + c] : ysB[(size_t)r*HH + (c - HH)];
    else
        v = 0.0f;
    out[i] = (_Float16)v;
}

// ---------------- kernel 6: top-q pooling + head (one block per batch row) -----------
// final[b] = (sum over top-k rows by ||out[b,t,:]|| of  out[b,t,:].Wc) / k + bc
__global__ void pool_topq(const _Float16* __restrict__ out, const int* __restrict__ lengths,
                          const float* __restrict__ Wc, const float* __restrict__ bc,
                          float* __restrict__ result){
    __shared__ float norms[TT];
    __shared__ float sv[TT];
    __shared__ float wcs[2*HH];
    __shared__ float rv[256];
    __shared__ int   ri[256];

    int b = blockIdx.x;
    int tid = threadIdx.x;
    int len = lengths[b];

    for (int c = tid; c < 2*HH; c += 256) wcs[c] = Wc[c];
    __syncthreads();

    for (int t = tid; t < TT; t += 256){
        const _Float16* row = out + ((size_t)(b*TT + t))*(2*HH);
        float nn = 0.0f, ss = 0.0f;
        for (int c = 0; c < 2*HH; ++c){
            float v = (float)row[c];
            nn += v*v;
            ss += v*wcs[c];
        }
        norms[t] = (t < len) ? sqrtf(nn) : -1e9f;
        sv[t] = ss;
    }
    __syncthreads();

    int ntop = (int)ceilf(0.15f * (float)len);
    if (ntop < 1) ntop = 1;

    float acc = 0.0f;    // thread-0's running top-k sum (register, never clobbered)
    for (int it = 0; it < ntop; ++it){
        float bv = -1e38f; int bi = 0;
        for (int t = tid; t < TT; t += 256){
            float v = norms[t];
            if (v > bv){ bv = v; bi = t; }
        }
        rv[tid] = bv; ri[tid] = bi;
        __syncthreads();
        for (int s = 128; s > 0; s >>= 1){
            if (tid < s && rv[tid+s] > rv[tid]){ rv[tid] = rv[tid+s]; ri[tid] = ri[tid+s]; }
            __syncthreads();
        }
        int sel = ri[0];
        __syncthreads();
        if (tid == 0){
            acc += sv[sel];
            norms[sel] = -1e30f;     // remove from candidate set
        }
        __syncthreads();
    }
    if (tid == 0) result[b] = acc / (float)ntop + bc[0];
}

// ---------------- host orchestration -------------------------------------------------
extern "C" void kernel_launch(void* const* d_in, const int* in_sizes, int n_in,
                              void* d_out, int out_size, void* d_ws, size_t ws_size,
                              hipStream_t stream){
    const float* raw    = (const float*)d_in[0];
    const int*   lens   = (const int*)  d_in[1];
    const float* ln_g   = (const float*)d_in[2];
    const float* ln_b   = (const float*)d_in[3];
    const float* W1     = (const float*)d_in[4];
    const float* b1     = (const float*)d_in[5];
    const float* W2     = (const float*)d_in[6];
    const float* b2     = (const float*)d_in[7];
    const float* Wih    = (const float*)d_in[8];
    const float* Whh    = (const float*)d_in[9];
    const float* bih    = (const float*)d_in[10];
    const float* bhh    = (const float*)d_in[11];
    const float* Wc     = (const float*)d_in[12];
    const float* bc     = (const float*)d_in[13];
    float* resOut = (float*)d_out;

    // workspace carve-out
    size_t off = 0;
    auto take = [&](size_t bytes)->char*{
        char* p = (char*)d_ws + off;
        off += (bytes + 255) & ~(size_t)255;
        return p;
    };
    _Float16* Xfeat = (_Float16*)take((size_t)MROWS*KPAD*2);
    _Float16* H1    = (_Float16*)take((size_t)MROWS*PROJ*2);
    _Float16* inpA  = (_Float16*)take((size_t)MROWS*PROJ*2);
    _Float16* inpB  = (_Float16*)take((size_t)MROWS*PROJ*2);
    float*    preF  = (float*)   take((size_t)MROWS*GG*4);
    float*    preB  = (float*)   take((size_t)MROWS*GG*4);
    float*    ysF   = (float*)   take((size_t)MROWS*HH*4);
    float*    ysB   = (float*)   take((size_t)MROWS*HH*4);
    _Float16* W1h   = (_Float16*)take((size_t)PROJ*KPAD*2);
    _Float16* W2h   = (_Float16*)take((size_t)PROJ*PROJ*2);
    _Float16* WihH  = (_Float16*)take((size_t)4*GG*PROJ*2);
    _Float16* WhhH  = (_Float16*)take((size_t)4*GG*HH*2);

    (void)in_sizes; (void)n_in; (void)out_size; (void)ws_size;

    hipFuncSetAttribute((const void*)gru_scan,
                        hipFuncAttributeMaxDynamicSharedMemorySize, (int)SMEM_GRU);

    // 1) features + LN
    features_ln<<<MROWS/256, 256, 0, stream>>>(raw, ln_g, ln_b, Xfeat);

    // 2) weight conversions to f16
    convert_pad<<<(PROJ*KPAD+255)/256, 256, 0, stream>>>(W1, W1h, PROJ, FEAT, KPAD);
    convert_pad<<<(PROJ*PROJ+255)/256, 256, 0, stream>>>(W2, W2h, PROJ, PROJ, PROJ);
    convert_pad<<<(4*GG*PROJ+255)/256, 256, 0, stream>>>(Wih, WihH, 4*GG, PROJ, PROJ);
    convert_pad<<<(4*GG*HH+255)/256, 256, 0, stream>>>(Whh, WhhH, 4*GG, HH, HH);

    // 3) MLP: gelu(X@W1^T+b1) -> gelu(.@W2^T+b2) * mask
    gemm_wmma<<<dim3(PROJ/64, MROWS/16), 128, 0, stream>>>(
        Xfeat, W1h, b1, nullptr, H1, MROWS, PROJ, KPAD, 1, nullptr, TT);
    gemm_wmma<<<dim3(PROJ/64, MROWS/16), 128, 0, stream>>>(
        H1, W2h, b2, nullptr, inpA, MROWS, PROJ, PROJ, 1, lens, TT);

    // 4) two GRU layers
    _Float16* inCur = inpA;
    _Float16* inNxt = inpB;
    for (int l = 0; l < 2; ++l){
        const _Float16* WihF16 = WihH + (size_t)(l*2+0)*GG*PROJ;
        const _Float16* WihB16 = WihH + (size_t)(l*2+1)*GG*PROJ;
        const _Float16* WhhF16 = WhhH + (size_t)(l*2+0)*GG*HH;
        const _Float16* WhhB16 = WhhH + (size_t)(l*2+1)*GG*HH;
        const float* bihF = bih + (size_t)(l*2+0)*GG;
        const float* bihB = bih + (size_t)(l*2+1)*GG;
        const float* bhhF = bhh + (size_t)(l*2+0)*GG;
        const float* bhhB = bhh + (size_t)(l*2+1)*GG;

        // pre_i = inp @ Wih^T + bih   (f32 out, no activation)
        gemm_wmma<<<dim3(GG/64, MROWS/16), 128, 0, stream>>>(
            inCur, WihF16, bihF, preF, nullptr, MROWS, GG, PROJ, 0, nullptr, TT);
        gemm_wmma<<<dim3(GG/64, MROWS/16), 128, 0, stream>>>(
            inCur, WihB16, bihB, preB, nullptr, MROWS, GG, PROJ, 0, nullptr, TT);

        // sequential scan over T, both directions concurrently
        gru_scan<<<2, 256, SMEM_GRU, stream>>>(
            preF, preB, WhhF16, WhhB16, bihF, bihB, bhhF, bhhB, ysF, ysB, lens);

        // concat + mask -> next-layer input (f16)
        concat_mask<<<(unsigned)(((size_t)MROWS*(2*HH))/256), 256, 0, stream>>>(
            ysF, ysB, lens, inNxt);

        _Float16* tmp = inCur; inCur = inNxt; inNxt = tmp;
    }

    // 5) top-q pooling + scalar head
    pool_topq<<<BB, 256, 0, stream>>>(inCur, lens, Wc, bc, resOut);
}
